// Gate_38671885533259
// MI455X (gfx1250) — matrix operand order repaired and checked by reading, not compiled, not run on hardware
//
#include <hip/hip_runtime.h>
#include <hip/hip_bf16.h>

#define TOKENS 16384
#define DIM 7168
#define NEXP 256
#define NGRP 8
#define GSIZE 32
#define TOPK_G 4
#define TOPK 8
#define ROUTE_SCALE 2.5f

#define KSEG 3584               // K-phase length (DIM/2): halves LDS -> 2 blocks/WGP
#define LDS_STRIDE 260          // score row stride (floats), pad vs 64-bank periodicity
#define ASTRIDE    3592         // bf16 A row stride (3584 + 8 pad), keeps 16B alignment

typedef __attribute__((ext_vector_type(16))) __bf16        v16bf;
typedef __attribute__((ext_vector_type(8)))  float         v8f;
typedef __attribute__((ext_vector_type(8)))  unsigned int  v8u;

// f32x2 -> packed bf16x2 via native conversion
__device__ __forceinline__ unsigned pk2c(float lo, float hi) {
    __bf16 a = (__bf16)lo;
    __bf16 b = (__bf16)hi;
    unsigned short ua = __builtin_bit_cast(unsigned short, a);
    unsigned short ub = __builtin_bit_cast(unsigned short, b);
    return (unsigned)ua | ((unsigned)ub << 16);
}

// ---- pre-pass: weight f32 -> bf16 (3.67 MB, runs once, stays L2-resident) ----
__global__ __launch_bounds__(256)
void convert_weight_kernel(const float* __restrict__ w, __bf16* __restrict__ o) {
    const size_t base = ((size_t)blockIdx.x * 256 + threadIdx.x) * 8;
    float4 f0 = *(const float4*)(w + base);
    float4 f1 = *(const float4*)(w + base + 4);
    uint4 u;
    u.x = pk2c(f0.x, f0.y); u.y = pk2c(f0.z, f0.w);
    u.z = pk2c(f1.x, f1.y); u.w = pk2c(f1.z, f1.w);
    *(uint4*)(o + base) = u;
}

union GateShMem {
    __bf16 abf[16 * ASTRIDE];     // GEMM phase: bf16 x K-slab (~112 KB)
    float  sc[16 * LDS_STRIDE];   // routing phase: sigmoid scores
};

template <bool USE_WBF>
__global__ __launch_bounds__(256)
void gate_moe_kernel(const float* __restrict__ x,
                     const float* __restrict__ weight,
                     const __bf16* __restrict__ wbf,
                     const float* __restrict__ bias,
                     float* __restrict__ out_w,
                     int*   __restrict__ out_i)
{
    __shared__ GateShMem sh;
    __shared__ float bsh[NEXP];

    const int tid  = threadIdx.x;
    const int wave = tid >> 5;
    const int lane = tid & 31;
    const int half = lane >> 4;
    const int l16  = lane & 15;
    const int t0   = blockIdx.x * 16;

    if (tid < NEXP) bsh[tid] = bias[tid];

    // Each wave owns two 16x16 E-tiles.
    const int et0 = wave * 2;
    v8f acc0 = {};
    v8f acc1 = {};

    const __bf16* abase = sh.abf + l16 * ASTRIDE;     // A row: M = l16
    const __bf16* wb0 = USE_WBF ? wbf + (size_t)(et0 * 16 + l16) * DIM : nullptr;
    const __bf16* wb1 = USE_WBF ? wbf + (size_t)((et0 + 1) * 16 + l16) * DIM : nullptr;
    const float* wf0 = weight + (size_t)(et0 * 16 + l16) * DIM;
    const float* wf1 = weight + (size_t)((et0 + 1) * 16 + l16) * DIM;

    // staging coordinates (fixed per thread)
    const int sm   = tid >> 4;          // token row 0..15
    const int sseg = tid & 15;          // 224-element segment of the K-phase

    for (int kp = 0; kp < DIM; kp += KSEG) {
        // ---- stage x[16 rows, kp..kp+KSEG) as bf16 into LDS ----
        {
            const float* __restrict__ src = x + (size_t)(t0 + sm) * DIM + kp + sseg * 224;
            __bf16* dst = sh.abf + sm * ASTRIDE + sseg * 224;
            #pragma unroll 4
            for (int i = 0; i < 224; i += 8) {
                float4 f0 = *(const float4*)(src + i);
                float4 f1 = *(const float4*)(src + i + 4);
                uint4 u;
                u.x = pk2c(f0.x, f0.y); u.y = pk2c(f0.z, f0.w);
                u.z = pk2c(f1.x, f1.y); u.w = pk2c(f1.z, f1.w);
                *(uint4*)(dst + i) = u;
            }
        }
        __syncthreads();

        const __bf16* pb0 = USE_WBF ? wb0 + kp : nullptr;
        const __bf16* pb1 = USE_WBF ? wb1 + kp : nullptr;
        const float*  pf0 = wf0 + kp;
        const float*  pf1 = wf1 + kp;

        #pragma unroll 4
        for (int k0 = 0; k0 < KSEG; k0 += 32) {
            // A fragment from LDS: 16-bit A 16x32 layout
            const int ka = k0 + half * 16;
            uint4 a0 = *(const uint4*)(abase + ka);        // K ka..ka+7
            uint4 a1 = *(const uint4*)(abase + ka + 8);    // K ka+8..ka+15
            v8u ap;
            ap[0] = a0.x; ap[1] = a0.y; ap[2] = a0.z; ap[3] = a0.w;
            ap[4] = a1.x; ap[5] = a1.y; ap[6] = a1.z; ap[7] = a1.w;
            v16bf afrag = __builtin_bit_cast(v16bf, ap);

            // B fragments: 16-bit B 32x16 layout, per-lane contiguous K range
            const int kb = k0 + half * 16;
            v8u bp0, bp1;
            if constexpr (USE_WBF) {
                uint4 q0 = *(const uint4*)(pb0 + kb);
                uint4 q1 = *(const uint4*)(pb0 + kb + 8);
                bp0[0] = q0.x; bp0[1] = q0.y; bp0[2] = q0.z; bp0[3] = q0.w;
                bp0[4] = q1.x; bp0[5] = q1.y; bp0[6] = q1.z; bp0[7] = q1.w;
                uint4 r0 = *(const uint4*)(pb1 + kb);
                uint4 r1 = *(const uint4*)(pb1 + kb + 8);
                bp1[0] = r0.x; bp1[1] = r0.y; bp1[2] = r0.z; bp1[3] = r0.w;
                bp1[4] = r1.x; bp1[5] = r1.y; bp1[6] = r1.z; bp1[7] = r1.w;
            } else {
                float4 b0 = *(const float4*)(pf0 + kb);
                float4 b1 = *(const float4*)(pf0 + kb + 4);
                float4 b2 = *(const float4*)(pf0 + kb + 8);
                float4 b3 = *(const float4*)(pf0 + kb + 12);
                bp0[0] = pk2c(b0.x, b0.y); bp0[1] = pk2c(b0.z, b0.w);
                bp0[2] = pk2c(b1.x, b1.y); bp0[3] = pk2c(b1.z, b1.w);
                bp0[4] = pk2c(b2.x, b2.y); bp0[5] = pk2c(b2.z, b2.w);
                bp0[6] = pk2c(b3.x, b3.y); bp0[7] = pk2c(b3.z, b3.w);
                float4 c0 = *(const float4*)(pf1 + kb);
                float4 c1 = *(const float4*)(pf1 + kb + 4);
                float4 c2 = *(const float4*)(pf1 + kb + 8);
                float4 c3 = *(const float4*)(pf1 + kb + 12);
                bp1[0] = pk2c(c0.x, c0.y); bp1[1] = pk2c(c0.z, c0.w);
                bp1[2] = pk2c(c1.x, c1.y); bp1[3] = pk2c(c1.z, c1.w);
                bp1[4] = pk2c(c2.x, c2.y); bp1[5] = pk2c(c2.z, c2.w);
                bp1[6] = pk2c(c3.x, c3.y); bp1[7] = pk2c(c3.z, c3.w);
            }
            v16bf bf0 = __builtin_bit_cast(v16bf, bp0);
            v16bf bf1 = __builtin_bit_cast(v16bf, bp1);
            acc0 = __builtin_amdgcn_wmma_f32_16x16x32_bf16(
                       false, afrag, false, bf0, (short)0, acc0, false, false);
            acc1 = __builtin_amdgcn_wmma_f32_16x16x32_bf16(
                       false, afrag, false, bf1, (short)0, acc1, false, false);
        }
        __syncthreads();   // A slab consumed; safe to restage / repurpose
    }

    // C/D layout: VGPR r, lane: M = r + 8*half, N = l16
    #pragma unroll
    for (int r = 0; r < 8; ++r) {
        const int m = r + 8 * half;
        sh.sc[m * LDS_STRIDE + et0 * 16 + l16]       = 1.f / (1.f + __expf(-acc0[r]));
        sh.sc[m * LDS_STRIDE + (et0 + 1) * 16 + l16] = 1.f / (1.f + __expf(-acc1[r]));
    }
    __syncthreads();

    // ---- routing, 16 lanes, one token each (cold tail) ----
    if (tid < 16) {
        const int row = tid;
        const float* srow = &sh.sc[row * LDS_STRIDE];

        float gs[NGRP];
        #pragma unroll
        for (int g = 0; g < NGRP; ++g) {
            float m1 = -1e30f, m2 = -1e30f;
            for (int j = 0; j < GSIZE; ++j) {
                float v = srow[g * GSIZE + j] + bsh[g * GSIZE + j];
                if (v > m1) { m2 = m1; m1 = v; }
                else if (v > m2) { m2 = v; }
            }
            gs[g] = m1 + m2;
        }

        unsigned gmask = 0;
        #pragma unroll
        for (int it = 0; it < TOPK_G; ++it) {
            float best = -1e30f; int bi = 0;
            for (int g = 0; g < NGRP; ++g)
                if (!((gmask >> g) & 1u) && gs[g] > best) { best = gs[g]; bi = g; }
            gmask |= 1u << bi;
        }

        unsigned chosen[NEXP / 32] = {0, 0, 0, 0, 0, 0, 0, 0};
        float wsel[TOPK]; int isel[TOPK];
        float wsum = 0.f;
        for (int it = 0; it < TOPK; ++it) {
            float best = -1e30f; int be = 0;
            for (int e = 0; e < NEXP; ++e) {
                if ((chosen[e >> 5] >> (e & 31)) & 1u) continue;
                const int g = e >> 5;
                float v = ((gmask >> g) & 1u) ? (srow[e] + bsh[e]) : 0.f;
                if (v > best) { best = v; be = e; }
            }
            chosen[be >> 5] |= 1u << (be & 31);
            isel[it] = be;
            wsel[it] = srow[be];              // ORIGINAL sigmoid score
            wsum += wsel[it];
        }

        const float scale = ROUTE_SCALE / wsum;
        const size_t base = (size_t)(t0 + row) * TOPK;
        #pragma unroll
        for (int it = 0; it < TOPK; ++it) {
            out_w[base + it] = wsel[it] * scale;
            out_i[base + it] = isel[it];
        }
    }
}

extern "C" void kernel_launch(void* const* d_in, const int* in_sizes, int n_in,
                              void* d_out, int out_size, void* d_ws, size_t ws_size,
                              hipStream_t stream) {
    const float* x      = (const float*)d_in[0];
    const float* weight = (const float*)d_in[1];
    const float* bias   = (const float*)d_in[2];
    float* out_w = (float*)d_out;                         // [T, 8] float32
    int*   out_i = (int*)d_out + (size_t)TOKENS * TOPK;   // [T, 8] int32

    const size_t WBYTES = (size_t)NEXP * DIM * sizeof(unsigned short);
    if (ws_size >= WBYTES) {
        __bf16* wbf = (__bf16*)d_ws;
        convert_weight_kernel<<<(NEXP * DIM) / (256 * 8), 256, 0, stream>>>(weight, wbf);
        gate_moe_kernel<true><<<TOKENS / 16, 256, 0, stream>>>(
            x, weight, wbf, bias, out_w, out_i);
    } else {
        gate_moe_kernel<false><<<TOKENS / 16, 256, 0, stream>>>(
            x, weight, nullptr, bias, out_w, out_i);
    }
}
